// OutputBlock_36352603194143
// MI455X (gfx1250) — matrix operand order repaired and checked by reading, not compiled, not run on hardware
//
#include <hip/hip_runtime.h>
#include <hip/hip_bf16.h>

// ---------------------------------------------------------------------------
// DimeNet-style output block for MI455X (gfx1250, wave32, WMMA).
//   Phase 1: zero node accumulator (25.6MB in d_ws)
//   Phase 2: edge kernel, m = (rbf@W_rbf.T)*x, scatter-add via HW f32 atomics
//            (h is L2-resident; x streamed non-temporally; per-instruction
//            lanes cover contiguous 128B so atomics coalesce per cacheline)
//   Phase 3: fused 4-layer MLP, f16 WMMA w/ f32 accumulation, activations +
//            weights LDS-resident per 64-row block; B frags held in regs,
//            A frags batch-loaded per tile, fast rcp-based silu epilogue.
// ---------------------------------------------------------------------------

typedef float    v4f  __attribute__((ext_vector_type(4)));
typedef float    v8f  __attribute__((ext_vector_type(8)));
typedef _Float16 v4h  __attribute__((ext_vector_type(4)));
typedef _Float16 v8h  __attribute__((ext_vector_type(8)));
typedef _Float16 v16h __attribute__((ext_vector_type(16)));

#define HIDDEN 128
#define NRAD   6
#define ROWS   64    // rows (nodes) per MLP block
#define LDH    136   // padded LDS row stride in halves (breaks bank alignment)

// Guaranteed single-instruction HW atomic (no CAS fallback).
__device__ __forceinline__ void atomic_add_f32(float* p, float v) {
    asm volatile("global_atomic_add_f32 %0, %1, off" : : "v"(p), "v"(v) : "memory");
}

// silu via v_exp_f32 + v_rcp_f32 (no IEEE divide expansion)
__device__ __forceinline__ float fast_silu(float v) {
    const float e = __builtin_amdgcn_exp2f(v * -1.442695040888963f);
    return v * __builtin_amdgcn_rcpf(1.f + e);
}

// ---------------------------------------------------------------------------
__global__ __launch_bounds__(256) void zero_ws(v4f* __restrict__ p, int n4) {
    int i = blockIdx.x * blockDim.x + threadIdx.x;
    if (i < n4) {
        v4f z = {0.f, 0.f, 0.f, 0.f};
        p[i] = z;
    }
}

// ---------------------------------------------------------------------------
// One wave per edge. Lane l owns channels {l, l+32, l+64, l+96}: each of the
// 4 load/atomic instructions touches 128 contiguous bytes across the wave.
__global__ __launch_bounds__(256) void edge_scatter(
    const float* __restrict__ x,     // [E,128]
    const float* __restrict__ rbf,   // [E,6]
    const int*   __restrict__ idx,   // [E]
    const float* __restrict__ Wrbf,  // [128,6]
    float*       __restrict__ h,     // [N,128] accumulator
    int E) {
    const int lane = threadIdx.x & 31;
    const int wid  = (blockIdx.x * blockDim.x + threadIdx.x) >> 5;
    const int nwav = (gridDim.x * blockDim.x) >> 5;

    float wr[4][NRAD];
#pragma unroll
    for (int j = 0; j < 4; ++j)
#pragma unroll
        for (int r = 0; r < NRAD; ++r)
            wr[j][r] = Wrbf[(lane + 32 * j) * NRAD + r];

    for (int e = wid; e < E; e += nwav) {
        const int node = idx[e];
        // rbf[e, 0..5]: three aligned float2 loads (wave-uniform address)
        const float2* rp = (const float2*)(rbf + (size_t)e * NRAD);
        const float2 rA = rp[0], rB = rp[1], rC = rp[2];
        const float rb[NRAD] = {rA.x, rA.y, rB.x, rB.y, rC.x, rC.y};

        const float* xrow = x + (size_t)e * HIDDEN + lane;
        float* hrow = h + (size_t)node * HIDDEN + lane;
#pragma unroll
        for (int j = 0; j < 4; ++j) {
            // streaming read of x: non-temporal, keep L2 for h + weights
            const float xv = __builtin_nontemporal_load(xrow + 32 * j);
            float coef = 0.f;
#pragma unroll
            for (int r = 0; r < NRAD; ++r)
                coef = fmaf(rb[r], wr[j][r], coef);
            atomic_add_f32(hrow + 32 * j, coef * xv);
        }
    }
}

// ---------------------------------------------------------------------------
// Fused 4-layer MLP. 8 waves / block; wave w owns output-column tile
// [16w, 16w+16). 64 rows per block; activations ping-pong in LDS as f16.
__global__ __launch_bounds__(256) void node_mlp(
    const float* __restrict__ hin,
    const float* __restrict__ W1, const float* __restrict__ b1,
    const float* __restrict__ W2, const float* __restrict__ b2,
    const float* __restrict__ W3, const float* __restrict__ b3,
    const float* __restrict__ Wout,
    float* __restrict__ out, int N) {
    __shared__ _Float16 actA[ROWS * LDH];
    __shared__ _Float16 actB[ROWS * LDH];
    __shared__ _Float16 Wl[HIDDEN * LDH];

    const int tid     = threadIdx.x;
    const int lane    = tid & 31;
    const int w       = tid >> 5;        // wave -> N tile
    const int lo      = lane & 15;
    const int hi      = lane >> 4;
    const int col     = (w << 4) + lo;   // this lane's output column
    const int rowBase = blockIdx.x * ROWS;

    // stage input activations f32 -> f16 (rows beyond N -> 0)
    for (int k = tid; k < ROWS * HIDDEN / 4; k += 256) {
        const int r = k >> 5;
        const int c = (k & 31) * 4;
        const int gr = rowBase + r;
        v4f v = {0.f, 0.f, 0.f, 0.f};
        if (gr < N) v = *(const v4f*)(hin + (size_t)gr * HIDDEN + c);
        v4h hv = {(_Float16)v[0], (_Float16)v[1], (_Float16)v[2], (_Float16)v[3]};
        *(v4h*)&actA[r * LDH + c] = hv;
    }

    const float* Ws[4] = {W1, W2, W3, Wout};
    const float* bs[4] = {b1, b2, b3, nullptr};
    _Float16* actIn  = actA;
    _Float16* actOut = actB;

#pragma unroll
    for (int L = 0; L < 4; ++L) {
        __syncthreads();  // prior layer done reading Wl / writing actOut
        // stage this layer's weights f32 -> f16 into LDS
        const float* W = Ws[L];
        for (int k = tid; k < HIDDEN * HIDDEN / 4; k += 256) {
            const int n = k >> 5;
            const int c = (k & 31) * 4;
            const v4f v = *(const v4f*)(W + n * HIDDEN + c);
            v4h hv = {(_Float16)v[0], (_Float16)v[1], (_Float16)v[2], (_Float16)v[3]};
            *(v4h*)&Wl[n * LDH + c] = hv;
        }
        __syncthreads();

        const float bcol = (L < 3) ? bs[L][col] : 0.f;

        // hoist B fragments into registers (depend on kc only, not on tile t)
        // B layout: lane = col, half j -> K = kc + hi*16 + j
        v16h bv[4];
#pragma unroll
        for (int kci = 0; kci < 4; ++kci) {
            const _Float16* bp = &Wl[col * LDH + kci * 32 + hi * 16];
            const v8h b0 = *(const v8h*)(bp);
            const v8h b1v = *(const v8h*)(bp + 8);
#pragma unroll
            for (int j = 0; j < 8; ++j) {
                bv[kci][j] = b0[j];
                bv[kci][j + 8] = b1v[j];
            }
        }

        const int kb = hi * 16;  // A frag K base (contiguous, matches B)
#pragma unroll
        for (int t = 0; t < 4; ++t) {  // 4 row tiles of 16
            const _Float16* arow = &actIn[(t * 16 + lo) * LDH];
            // batch-load all A fragments of this tile first so the
            // s_wait_dscnt before each WMMA can stagger (6/4/2/0)
            v16h av[4];
#pragma unroll
            for (int kci = 0; kci < 4; ++kci) {
                const int kc = kci * 32;
                // A: lane = row, half j -> K = kc + hi*16 + j
                const v8h a0 = *(const v8h*)(arow + kc + kb);
                const v8h a1 = *(const v8h*)(arow + kc + kb + 8);
#pragma unroll
                for (int j = 0; j < 8; ++j) {
                    av[kci][j] = a0[j];
                    av[kci][j + 8] = a1[j];
                }
            }
            v8f acc = {};
#pragma unroll
            for (int kci = 0; kci < 4; ++kci)
                acc = __builtin_amdgcn_wmma_f32_16x16x32_f16(
                    false, av[kci], false, bv[kci], (short)0, acc, false, false);

            if (L < 3) {
                // bias + silu, back to f16 LDS (C layout: row = t*16+hi*8+r)
#pragma unroll
                for (int r = 0; r < 8; ++r) {
                    const float v = fast_silu(acc[r] + bcol);
                    actOut[(t * 16 + hi * 8 + r) * LDH + col] = (_Float16)v;
                }
            } else {
#pragma unroll
                for (int r = 0; r < 8; ++r) {
                    const int gr = rowBase + t * 16 + hi * 8 + r;
                    if (gr < N) out[(size_t)gr * HIDDEN + col] = acc[r];
                }
            }
        }
        _Float16* tmp = actIn; actIn = actOut; actOut = tmp;
    }
}

// ---------------------------------------------------------------------------
extern "C" void kernel_launch(void* const* d_in, const int* in_sizes, int n_in,
                              void* d_out, int out_size, void* d_ws, size_t ws_size,
                              hipStream_t stream) {
    const float* x    = (const float*)d_in[0];
    const float* rbf  = (const float*)d_in[1];
    const int*   idx  = (const int*)d_in[2];
    // d_in[3] = num_nodes scalar on device; N derived from out_size instead
    const float* Wrbf = (const float*)d_in[4];
    const float* W1   = (const float*)d_in[5];
    const float* b1   = (const float*)d_in[6];
    const float* W2   = (const float*)d_in[7];
    const float* b2   = (const float*)d_in[8];
    const float* W3   = (const float*)d_in[9];
    const float* b3   = (const float*)d_in[10];
    const float* Wout = (const float*)d_in[11];

    const int E = in_sizes[0] / HIDDEN;
    const int N = out_size / HIDDEN;
    float* h = (float*)d_ws;  // [N,128] f32 accumulator

    const int n4 = N * (HIDDEN / 4);
    zero_ws<<<(n4 + 255) / 256, 256, 0, stream>>>((v4f*)h, n4);

    // 8192 blocks * 8 waves = 64k waves: enough concurrency for 23.3 TB/s
    edge_scatter<<<8192, 256, 0, stream>>>(x, rbf, idx, Wrbf, h, E);

    node_mlp<<<(N + ROWS - 1) / ROWS, 256, 0, stream>>>(
        h, W1, b1, W2, b2, W3, b3, Wout, (float*)d_out, N);
}
